// HSL_78486232367137
// MI455X (gfx1250) — compile-verified, matrix operand order checked by reading, not run on hardware
//
#include <hip/hip_runtime.h>
#include <hip/hip_bf16.h>
#include <stdint.h>

#define NN 100000
#define EE 1600000
#define MM 50000

typedef __attribute__((ext_vector_type(8)))  float          v8f;
typedef __attribute__((ext_vector_type(16))) __bf16         v16bf;
typedef __attribute__((ext_vector_type(16))) unsigned short u16x16;

__device__ __forceinline__ unsigned short f2bf(float f) {
  unsigned u = __float_as_uint(f);
  u += 0x7FFFu + ((u >> 16) & 1u);          // round-to-nearest-even
  return (unsigned short)(u >> 16);
}
__device__ __forceinline__ float bf2f(unsigned short s) {
  return __uint_as_float(((unsigned)s) << 16);
}
// monotone float <-> uint encoding for atomicMax
__device__ __forceinline__ unsigned encf(float f) {
  unsigned u = __float_as_uint(f);
  return (u & 0x80000000u) ? ~u : (u | 0x80000000u);
}
__device__ __forceinline__ float decf(unsigned e) {
  unsigned u = (e & 0x80000000u) ? (e & 0x7FFFFFFFu) : ~e;
  return __uint_as_float(u);
}

// ---- split fp32 weights into bf16 hi/lo -----------------------------------
__global__ __launch_bounds__(256) void k_wconv(const float* __restrict__ w,
                                               unsigned short* __restrict__ hi,
                                               unsigned short* __restrict__ lo, int n) {
  int i = blockIdx.x * 256 + threadIdx.x;
  if (i >= n) return;
  float f = w[i];
  unsigned short h = f2bf(f);
  hi[i] = h;
  lo[i] = f2bf(f - bf2f(h));
}

// ---- fold w_K / att_r into U[128][4], c[4] --------------------------------
__global__ __launch_bounds__(128) void k_prepU(const float* __restrict__ wK,
                                               const float* __restrict__ bK,
                                               const float* __restrict__ attr,
                                               float* __restrict__ U, float* __restrict__ cb) {
  int d = threadIdx.x;
  for (int h = 0; h < 4; ++h) {
    float s = 0.f;
    for (int c = 0; c < 32; ++c) s += wK[(h * 32 + c) * 128 + d] * attr[h * 32 + c];
    U[d * 4 + h] = s;
  }
  if (d < 4) {
    float s = 0.f;
    for (int c = 0; c < 32; ++c) s += bK[d * 32 + c] * attr[d * 32 + c];
    cb[d] = s;
  }
}

// ---- WMMA GEMM: Out[rows,128] = A[rows,128] @ W^T + bias, bf16-split ------
__global__ __launch_bounds__(256) void k_gemm128(const float* __restrict__ A,
                                                 const unsigned short* __restrict__ Whi,
                                                 const unsigned short* __restrict__ Wlo,
                                                 const float* __restrict__ bias,
                                                 float* __restrict__ Out, int rows, int act) {
  __shared__ __align__(16) unsigned short sHi[128 * 128];
  __shared__ __align__(16) unsigned short sLo[128 * 128];
  int tid = threadIdx.x;
  for (int i = tid; i < 2048; i += 256) {          // 2048 uint4 = 32KB per array
    ((uint4*)sHi)[i] = ((const uint4*)Whi)[i];
    ((uint4*)sLo)[i] = ((const uint4*)Wlo)[i];
  }
  __syncthreads();

  int lane = tid & 31;
  int wave = tid >> 5;
  int row0 = (blockIdx.x * 8 + wave) * 16;
  if (row0 >= rows) return;

  int nlane = lane & 15;
  int kbase = (lane & 16) ? 8 : 0;                 // ISA 16-bit A/B half->K mapping
  int kk[16];
#pragma unroll
  for (int h = 0; h < 16; ++h) kk[h] = h + (h & 8) + kbase;

  const float* Arow = A + (size_t)(row0 + nlane) * 128;

  v8f acc[8];
#pragma unroll
  for (int n = 0; n < 8; ++n) {
    float bv = bias[n * 16 + nlane];
#pragma unroll
    for (int i = 0; i < 8; ++i) acc[n][i] = bv;
  }

#pragma unroll
  for (int kt = 0; kt < 4; ++kt) {
    u16x16 ah, al;
#pragma unroll
    for (int h = 0; h < 16; ++h) {
      float v = Arow[kt * 32 + kk[h]];
      unsigned short hb = f2bf(v);
      ah[h] = hb;
      al[h] = f2bf(v - bf2f(hb));
    }
    v16bf Ah = __builtin_bit_cast(v16bf, ah);
    v16bf Al = __builtin_bit_cast(v16bf, al);
#pragma unroll
    for (int n = 0; n < 8; ++n) {
      int base = (n * 16 + nlane) * 128 + kt * 32;
      u16x16 bh, bl;
#pragma unroll
      for (int h = 0; h < 16; ++h) { bh[h] = sHi[base + kk[h]]; bl[h] = sLo[base + kk[h]]; }
      v16bf Bh = __builtin_bit_cast(v16bf, bh);
      v16bf Bl = __builtin_bit_cast(v16bf, bl);
      acc[n] = __builtin_amdgcn_wmma_f32_16x16x32_bf16(false, Ah, false, Bh, (short)0, acc[n], false, false);
      acc[n] = __builtin_amdgcn_wmma_f32_16x16x32_bf16(false, Ah, false, Bl, (short)0, acc[n], false, false);
      acc[n] = __builtin_amdgcn_wmma_f32_16x16x32_bf16(false, Al, false, Bh, (short)0, acc[n], false, false);
    }
  }

  int mbase = row0 + ((lane >> 4) << 3);
#pragma unroll
  for (int n = 0; n < 8; ++n) {
#pragma unroll
    for (int v = 0; v < 8; ++v) {
      float val = acc[n][v];
      if (act) val = fmaxf(val, 0.f);
      Out[(size_t)(mbase + v) * 128 + n * 16 + nlane] = val;
    }
  }
}

// ---- alpha_r = leaky(x @ U + c) -------------------------------------------
__global__ __launch_bounds__(256) void k_alpha(const float* __restrict__ x,
                                               const float* __restrict__ U,
                                               const float* __restrict__ cb,
                                               float* __restrict__ aN) {
  int i = blockIdx.x * 256 + threadIdx.x;
  if (i >= NN * 4) return;
  int n = i >> 2, h = i & 3;
  const float* xr = x + (size_t)n * 128;
  float s = cb[h];
  for (int d = 0; d < 128; ++d) s += xr[d] * U[d * 4 + h];
  aN[i] = (s > 0.f) ? s : 0.2f * s;
}

// ---- init: agg = att_r broadcast; max/den = identity ----------------------
__global__ __launch_bounds__(256) void k_init(float* __restrict__ agg, unsigned* __restrict__ segMax,
                                              float* __restrict__ segDen, const float* __restrict__ attr) {
  int i = blockIdx.x * 256 + threadIdx.x;
  if (i < MM * 128) agg[i] = attr[i & 127];
  if (i < MM * 4) { segMax[i] = 0u; segDen[i] = 0.f; }
}

__global__ __launch_bounds__(256) void k_segmax(const int* __restrict__ src, const int* __restrict__ seg,
                                                const float* __restrict__ aN, unsigned* __restrict__ segMax) {
  int i = blockIdx.x * 256 + threadIdx.x;
  if (i >= EE * 4) return;
  int e = i >> 2, h = i & 3;
  atomicMax(&segMax[seg[e] * 4 + h], encf(aN[src[e] * 4 + h]));
}

__global__ __launch_bounds__(256) void k_segden(const int* __restrict__ src, const int* __restrict__ seg,
                                                const float* __restrict__ aN, const unsigned* __restrict__ segMax,
                                                float* __restrict__ segDen) {
  int i = blockIdx.x * 256 + threadIdx.x;
  if (i >= EE * 4) return;
  int e = i >> 2, h = i & 3;
  int s = seg[e] * 4 + h;
  atomicAdd(&segDen[s], __expf(aN[src[e] * 4 + h] - decf(segMax[s])));
}

// ---- weighted scatter-add: one wave per edge, float4 per lane -------------
__global__ __launch_bounds__(256) void k_scatter(const int* __restrict__ src_idx, const int* __restrict__ seg_idx,
                                                 const float* __restrict__ aN, const unsigned* __restrict__ segMax,
                                                 const float* __restrict__ segDen, const float* __restrict__ xV,
                                                 float* __restrict__ agg) {
  int lane = threadIdx.x & 31;
  int e = blockIdx.x * 8 + (threadIdx.x >> 5);
  if (e >= EE) return;
  int src = src_idx[e], seg = seg_idx[e];
  int d0 = lane << 2;
  if (e + 8 < EE) __builtin_prefetch(&xV[(size_t)src_idx[e + 8] * 128 + d0], 0, 1);
  int h = d0 >> 5;
  int s4 = seg * 4 + h;
  float al = __expf(aN[src * 4 + h] - decf(segMax[s4])) / (segDen[s4] + 1e-16f);
  const float* vp = xV + (size_t)src * 128 + d0;
  float* dp = agg + (size_t)seg * 128 + d0;
  atomicAdd(dp + 0, vp[0] * al);
  atomicAdd(dp + 1, vp[1] * al);
  atomicAdd(dp + 2, vp[2] * al);
  atomicAdd(dp + 3, vp[3] * al);
}

// ---- LayerNorm (mode0: z=a ; mode1: z=a+relu(b)), one wave per row --------
__global__ __launch_bounds__(256) void k_ln(const float* __restrict__ a, const float* __restrict__ b,
                                            const float* __restrict__ g, const float* __restrict__ bb,
                                            float* __restrict__ out, int rows, int mode) {
  int lane = threadIdx.x & 31;
  int row = blockIdx.x * 8 + (threadIdx.x >> 5);
  if (row >= rows) return;
  size_t base = (size_t)row * 128 + lane * 4;
  float z[4];
#pragma unroll
  for (int i = 0; i < 4; ++i) {
    float v = a[base + i];
    if (mode) v += fmaxf(b[base + i], 0.f);
    z[i] = v;
  }
  float s = z[0] + z[1] + z[2] + z[3];
  for (int m = 16; m > 0; m >>= 1) s += __shfl_xor(s, m, 32);
  float mu = s * (1.f / 128.f);
  float q = 0.f;
#pragma unroll
  for (int i = 0; i < 4; ++i) { float d = z[i] - mu; q += d * d; }
  for (int m = 16; m > 0; m >>= 1) q += __shfl_xor(q, m, 32);
  float inv = rsqrtf(q * (1.f / 128.f) + 1e-5f);
#pragma unroll
  for (int i = 0; i < 4; ++i) {
    int col = lane * 4 + i;
    out[base + i] = (z[i] - mu) * inv * g[col] + bb[col];
  }
}

extern "C" void kernel_launch(void* const* d_in, const int* in_sizes, int n_in,
                              void* d_out, int out_size, void* d_ws, size_t ws_size,
                              hipStream_t stream) {
  (void)in_sizes; (void)n_in; (void)out_size; (void)ws_size;
  const float* x    = (const float*)d_in[0];
  const int*   srcI = (const int*)d_in[1];
  const int*   segI = (const int*)d_in[2];
  const float* wK   = (const float*)d_in[3];
  const float* bK   = (const float*)d_in[4];
  const float* wV   = (const float*)d_in[5];
  const float* bV   = (const float*)d_in[6];
  const float* attr = (const float*)d_in[7];
  const float* ln0g = (const float*)d_in[8];
  const float* ln0b = (const float*)d_in[9];
  const float* ln1g = (const float*)d_in[10];
  const float* ln1b = (const float*)d_in[11];
  const float* w1   = (const float*)d_in[12];
  const float* b1   = (const float*)d_in[13];
  const float* w2   = (const float*)d_in[14];
  const float* b2   = (const float*)d_in[15];

  char* w = (char*)d_ws;
  size_t off = 0;
  float*    xV     = (float*)(w + off);    off += (size_t)NN * 128 * 4;  // 51.2 MB
  float*    aN     = (float*)(w + off);    off += (size_t)NN * 4 * 4;
  unsigned* segMax = (unsigned*)(w + off); off += (size_t)MM * 4 * 4;
  float*    segDen = (float*)(w + off);    off += (size_t)MM * 4 * 4;
  float*    agg    = (float*)(w + off);    off += (size_t)MM * 128 * 4;  // also reused as y
  float*    outln  = (float*)(w + off);    off += (size_t)MM * 128 * 4;
  float*    hbuf   = (float*)(w + off);    off += (size_t)MM * 128 * 4;
  unsigned short* wVhi = (unsigned short*)(w + off); off += 128 * 128 * 2;
  unsigned short* wVlo = (unsigned short*)(w + off); off += 128 * 128 * 2;
  unsigned short* w1hi = (unsigned short*)(w + off); off += 128 * 128 * 2;
  unsigned short* w1lo = (unsigned short*)(w + off); off += 128 * 128 * 2;
  unsigned short* w2hi = (unsigned short*)(w + off); off += 128 * 128 * 2;
  unsigned short* w2lo = (unsigned short*)(w + off); off += 128 * 128 * 2;
  float* U  = (float*)(w + off); off += 128 * 4 * 4;
  float* cb = (float*)(w + off); off += 16;

  k_wconv<<<64, 256, 0, stream>>>(wV, wVhi, wVlo, 128 * 128);
  k_wconv<<<64, 256, 0, stream>>>(w1, w1hi, w1lo, 128 * 128);
  k_wconv<<<64, 256, 0, stream>>>(w2, w2hi, w2lo, 128 * 128);
  k_prepU<<<1, 128, 0, stream>>>(wK, bK, attr, U, cb);

  k_gemm128<<<782, 256, 0, stream>>>(x, wVhi, wVlo, bV, xV, NN, 0);      // x_V
  k_alpha<<<(NN * 4 + 255) / 256, 256, 0, stream>>>(x, U, cb, aN);

  k_init<<<(MM * 128 + 255) / 256, 256, 0, stream>>>(agg, segMax, segDen, attr);
  k_segmax<<<(EE * 4 + 255) / 256, 256, 0, stream>>>(srcI, segI, aN, segMax);
  k_segden<<<(EE * 4 + 255) / 256, 256, 0, stream>>>(srcI, segI, aN, segMax, segDen);
  k_scatter<<<EE / 8, 256, 0, stream>>>(srcI, segI, aN, segMax, segDen, xV, agg);

  k_ln<<<(MM + 7) / 8, 256, 0, stream>>>(agg, nullptr, ln0g, ln0b, outln, MM, 0);
  k_gemm128<<<391, 256, 0, stream>>>(outln, w1hi, w1lo, b1, hbuf, MM, 1); // relu
  k_gemm128<<<391, 256, 0, stream>>>(hbuf, w2hi, w2lo, b2, agg, MM, 0);   // y (reuse agg)
  k_ln<<<(MM + 7) / 8, 256, 0, stream>>>(outln, agg, ln1g, ln1b, (float*)d_out, MM, 1);
}
